// RGCN_21526376088370
// MI455X (gfx1250) — matrix operand order repaired and checked by reading, not compiled, not run on hardware
//
#include <hip/hip_runtime.h>
#include <hip/hip_bf16.h>
#include <math.h>

// Problem constants (reference: BS,N,D,R = 16,512,256,2)
#define BSZ 16
#define NN  512
#define DDIM 256
#define JT  (NN / 16)    // 32 j-tiles
#define KT32 (NN / 32)   // 16 k-tiles over nodes
#define NT  (DDIM / 16)  // 16 feature tiles
#define KT8 (DDIM / 32)  // 8 k-tiles over features

typedef __attribute__((ext_vector_type(16))) __bf16 v16bf;
typedef __attribute__((ext_vector_type(8)))  __bf16 v8bf;
typedef __attribute__((ext_vector_type(8)))  float  v8f;

static __device__ __forceinline__ v8f wmma_bf16(v16bf a, v16bf b, v8f c) {
  // (neg_a, A, neg_b, B, c_mod, C, reuse_a, reuse_b)
  return __builtin_amdgcn_wmma_f32_16x16x32_bf16(false, a, false, b, (short)0, c,
                                                 false, false);
}

// Load a 16-half fragment stored contiguously per lane (two aligned b128s).
static __device__ __forceinline__ v16bf load_frag16(const __bf16* p) {
  v8bf x0 = *(const v8bf*)p;
  v8bf x1 = *(const v8bf*)(p + 8);
  v16bf r;
#pragma unroll
  for (int e = 0; e < 8; ++e) { r[e] = x0[e]; r[8 + e] = x1[e]; }
  return r;
}

// ---------------------------------------------------------------------------
// Kernel 1: per-relation in-degree of each target node j, inverted.
// invdeg layout: [r=0][BS*NN], [r=1][BS*NN]
// ---------------------------------------------------------------------------
__global__ void rgcn_deg_kernel(const int* __restrict__ aug,
                                const int* __restrict__ pun,
                                float* __restrict__ invdeg) {
  int idx = blockIdx.x * blockDim.x + threadIdx.x;
  if (idx >= BSZ * NN) return;
  int b = idx / NN, j = idx % NN;
  const int* ab = aug + (size_t)b * NN * NN + j;
  const int* pb = pun + (size_t)b * NN * NN + j;
  int c0 = 0, c1 = 0;
  for (int i = 0; i < NN; ++i) {
    int a = ab[(size_t)i * NN];
    int p = pb[(size_t)i * NN];
    c0 += (p == 1 && a != 1) ? 1 : 0;
    c1 += (a == 1) ? 1 : 0;
  }
  if (c0 < 1) c0 = 1;
  if (c1 < 1) c1 = 1;
  invdeg[idx]            = 1.0f / (float)c0;
  invdeg[BSZ * NN + idx] = 1.0f / (float)c1;
}

// ---------------------------------------------------------------------------
// Kernel 2: pack both relation masks into WMMA A-fragment layout (bf16).
// A-tile (b, jt, it): A[m,k] = mask_r[b, i = it*32+k, j = jt*16+m].
// Per-lane 16 halves stored contiguously (32B). r=1 block follows r=0 block.
// ---------------------------------------------------------------------------
__global__ __launch_bounds__(32) void rgcn_maskpack_kernel(
    const int* __restrict__ aug, const int* __restrict__ pun,
    __bf16* __restrict__ mpack) {
  int blk = blockIdx.x;
  int it = blk % KT32; blk /= KT32;
  int jt = blk % JT;   blk /= JT;
  int b  = blk;
  int lane = threadIdx.x;
  int m = lane & 15, lo = (lane >> 4) * 8;  // ISA 16-bit A layout
  int j = jt * 16 + m;
  const int* ab = aug + (size_t)b * NN * NN + j;
  const int* pb = pun + (size_t)b * NN * NN + j;
  v8bf a0lo, a0hi, a1lo, a1hi;
#pragma unroll
  for (int e = 0; e < 8; ++e) {
    int i0 = it * 32 + lo + e;        // fragment halves 0..7
    int i1 = it * 32 + 16 + lo + e;   // fragment halves 8..15
    int av0 = ab[(size_t)i0 * NN], pv0 = pb[(size_t)i0 * NN];
    int av1 = ab[(size_t)i1 * NN], pv1 = pb[(size_t)i1 * NN];
    a0lo[e] = (__bf16)((pv0 == 1 && av0 != 1) ? 1.0f : 0.0f);
    a0hi[e] = (__bf16)((pv1 == 1 && av1 != 1) ? 1.0f : 0.0f);
    a1lo[e] = (__bf16)((av0 == 1) ? 1.0f : 0.0f);
    a1hi[e] = (__bf16)((av1 == 1) ? 1.0f : 0.0f);
  }
  size_t tile = (((size_t)b * JT + jt) * KT32 + it);
  const size_t ROFF = (size_t)BSZ * JT * KT32;  // r=1 tile offset
  __bf16* d0 = mpack + (tile * 32 + lane) * 16;
  __bf16* d1 = mpack + ((ROFF + tile) * 32 + lane) * 16;
  *(v8bf*)d0 = a0lo; *(v8bf*)(d0 + 8) = a0hi;
  *(v8bf*)d1 = a1lo; *(v8bf*)(d1 + 8) = a1hi;
}

// ---------------------------------------------------------------------------
// Kernel 3: pack all six f32 weight matrices into B-fragment bf16 layout.
// B-tile (mi, kt, nt): lane n-column, K halves per lane-half.
// ---------------------------------------------------------------------------
__global__ __launch_bounds__(32) void rgcn_wpack_kernel(
    const float* w0, const float* w1, const float* w2, const float* w3,
    const float* w4, const float* w5, __bf16* __restrict__ wpack) {
  int blk = blockIdx.x;
  int nt = blk % NT;  blk /= NT;
  int kt = blk % KT8; blk /= KT8;
  int mi = blk;  // 0..5
  const float* W = (mi == 0) ? w0 : (mi == 1) ? w1 : (mi == 2) ? w2
                 : (mi == 3) ? w3 : (mi == 4) ? w4 : w5;
  int lane = threadIdx.x;
  int n = nt * 16 + (lane & 15);
  int khalf = (lane >> 4) * 8;
  v8bf lo8, hi8;
#pragma unroll
  for (int e = 0; e < 8; ++e) {
    lo8[e] = (__bf16)W[(size_t)(kt * 32 + khalf + e) * DDIM + n];
    hi8[e] = (__bf16)W[(size_t)(kt * 32 + 16 + khalf + e) * DDIM + n];
  }
  __bf16* dst = wpack + ((((size_t)mi * KT8 + kt) * NT + nt) * 32 + lane) * 16;
  *(v8bf*)dst = lo8;
  *(v8bf*)(dst + 8) = hi8;
}

// ---------------------------------------------------------------------------
// Kernel 4: pack a feature matrix [BS,NN,DDIM] (f32 or bf16) into B-fragment
// layout for the aggregation GEMMs (K = node index, N = feature index).
// ---------------------------------------------------------------------------
template <bool XF32>
__global__ __launch_bounds__(32) void rgcn_featpack_kernel(
    const void* __restrict__ src, __bf16* __restrict__ featp) {
  int blk = blockIdx.x;
  int nt = blk % NT;   blk /= NT;
  int kt = blk % KT32; blk /= KT32;
  int b  = blk;
  int lane = threadIdx.x;
  int n = nt * 16 + (lane & 15);
  int khalf = (lane >> 4) * 8;
  v8bf lo8, hi8;
#pragma unroll
  for (int e = 0; e < 8; ++e) {
    size_t r0 = ((size_t)b * NN + kt * 32 + khalf + e) * DDIM + n;
    size_t r1 = ((size_t)b * NN + kt * 32 + 16 + khalf + e) * DDIM + n;
    if (XF32) {
      lo8[e] = (__bf16)((const float*)src)[r0];
      hi8[e] = (__bf16)((const float*)src)[r1];
    } else {
      lo8[e] = ((const __bf16*)src)[r0];
      hi8[e] = ((const __bf16*)src)[r1];
    }
  }
  __bf16* dst = featp + ((((size_t)b * KT32 + kt) * NT + nt) * 32 + lane) * 16;
  *(v8bf*)dst = lo8;
  *(v8bf*)(dst + 8) = hi8;
}

// ---------------------------------------------------------------------------
// Kernel 5: masked mean-aggregation, both relations, 16(j) x 64(d) per wave.
// All fragments are pre-packed: per K-step = 12 b128 loads -> 8 WMMAs.
// ---------------------------------------------------------------------------
__global__ __launch_bounds__(32) void rgcn_agg_kernel(
    const __bf16* __restrict__ featp, const __bf16* __restrict__ mpack,
    const float* __restrict__ invdeg, __bf16* __restrict__ agg0,
    __bf16* __restrict__ agg1) {
  int blk = blockIdx.x;
  int dg = blk % (NT / 4); blk /= (NT / 4);  // group of 4 d-tiles
  int jt = blk % JT;       blk /= JT;
  int b  = blk;
  int lane = threadIdx.x;
  const size_t ROFF = (size_t)BSZ * JT * KT32;

  v8f acc0[4] = {};
  v8f acc1[4] = {};
  for (int kt = 0; kt < KT32; ++kt) {
    size_t mtile = (((size_t)b * JT + jt) * KT32 + kt);
    v16bf a0 = load_frag16(mpack + (mtile * 32 + lane) * 16);
    v16bf a1 = load_frag16(mpack + ((ROFF + mtile) * 32 + lane) * 16);
#pragma unroll
    for (int g = 0; g < 4; ++g) {
      int nt = dg * 4 + g;
      v16bf bf = load_frag16(
          featp + ((((size_t)b * KT32 + kt) * NT + nt) * 32 + lane) * 16);
      acc0[g] = wmma_bf16(a0, bf, acc0[g]);
      acc1[g] = wmma_bf16(a1, bf, acc1[g]);
    }
  }
  // Epilogue: D-frag m = v + 8*(lane>=16), n = lane&15. Scale by 1/deg.
  int hi = lane >> 4, n = lane & 15;
#pragma unroll
  for (int g = 0; g < 4; ++g) {
    int d = (dg * 4 + g) * 16 + n;
#pragma unroll
    for (int v = 0; v < 8; ++v) {
      int m = v + 8 * hi;
      int j = jt * 16 + m;
      float s0 = invdeg[(size_t)b * NN + j];
      float s1 = invdeg[(size_t)BSZ * NN + (size_t)b * NN + j];
      size_t off = ((size_t)b * NN + j) * DDIM + d;
      agg0[off] = (__bf16)(acc0[g][v] * s0);
      agg1[off] = (__bf16)(acc1[g][v] * s1);
    }
  }
}

// ---------------------------------------------------------------------------
// Kernel 6: fused combine GEMM + bias + ELU, 16(j) x 64(e) per wave.
// K = 3*256 over [x|agg0|agg1]; A rows contiguous in K (2 vector loads),
// B from packed weights (2 b128 per WMMA).
// ---------------------------------------------------------------------------
template <bool IN_F32, bool OUT_F32>
__global__ __launch_bounds__(32) void rgcn_combine_kernel(
    const void* __restrict__ xsrc, const __bf16* __restrict__ agg0,
    const __bf16* __restrict__ agg1, const __bf16* __restrict__ wpack,
    const float* __restrict__ bias, void* __restrict__ out) {
  int blk = blockIdx.x;
  int eg = blk % (NT / 4); blk /= (NT / 4);
  int jt = blk % JT;       blk /= JT;
  int b  = blk;
  int lane = threadIdx.x;
  int mrow = lane & 15, hi = lane >> 4, lo = hi * 8;
  int jA = jt * 16 + mrow;

  v8f acc[4] = {};
#pragma unroll
  for (int s = 0; s < 3; ++s) {
    for (int kt = 0; kt < KT8; ++kt) {
      int k0 = kt * 32;
      v16bf afrag;
      if (s == 0 && IN_F32) {
        const float* row = (const float*)xsrc + ((size_t)b * NN + jA) * DDIM;
        v8f f0 = *(const v8f*)(row + k0 + lo);
        v8f f1 = *(const v8f*)(row + k0 + 16 + lo);
#pragma unroll
        for (int e = 0; e < 8; ++e) {
          afrag[e] = (__bf16)f0[e];
          afrag[8 + e] = (__bf16)f1[e];
        }
      } else {
        const __bf16* row;
        if (s == 0)      row = (const __bf16*)xsrc + ((size_t)b * NN + jA) * DDIM;
        else if (s == 1) row = agg0 + ((size_t)b * NN + jA) * DDIM;
        else             row = agg1 + ((size_t)b * NN + jA) * DDIM;
        v8bf x0 = *(const v8bf*)(row + k0 + lo);
        v8bf x1 = *(const v8bf*)(row + k0 + 16 + lo);
#pragma unroll
        for (int e = 0; e < 8; ++e) { afrag[e] = x0[e]; afrag[8 + e] = x1[e]; }
      }
#pragma unroll
      for (int g = 0; g < 4; ++g) {
        int nt = eg * 4 + g;
        v16bf bfrag = load_frag16(
            wpack + ((((size_t)s * KT8 + kt) * NT + nt) * 32 + lane) * 16);
        acc[g] = wmma_bf16(afrag, bfrag, acc[g]);
      }
    }
  }
  int n = lane & 15;
#pragma unroll
  for (int g = 0; g < 4; ++g) {
    float bv = bias[(eg * 4 + g) * 16 + n];
#pragma unroll
    for (int v = 0; v < 8; ++v) {
      int m = v + 8 * hi;
      float val = acc[g][v] + bv;
      val = (val > 0.0f) ? val : (__expf(val) - 1.0f);  // ELU (alpha = 1)
      size_t off = ((size_t)b * NN + jt * 16 + m) * DDIM + (eg * 4 + g) * 16 + n;
      if (OUT_F32) ((float*)out)[off] = val;
      else         ((__bf16*)out)[off] = (__bf16)val;
    }
  }
}

// ---------------------------------------------------------------------------
extern "C" void kernel_launch(void* const* d_in, const int* in_sizes, int n_in,
                              void* d_out, int out_size, void* d_ws,
                              size_t ws_size, hipStream_t stream) {
  (void)in_sizes; (void)n_in; (void)out_size; (void)ws_size;
  const float* x       = (const float*)d_in[0];  // [BS,N,D]
  const float* w_rel1  = (const float*)d_in[1];  // [2,D,D]
  const float* w_root1 = (const float*)d_in[2];  // [D,D]
  const float* b1      = (const float*)d_in[3];  // [D]
  const float* w_rel2  = (const float*)d_in[4];
  const float* w_root2 = (const float*)d_in[5];
  const float* b2      = (const float*)d_in[6];
  const int*   aug     = (const int*)d_in[7];    // [BS,N,N]
  const int*   pun     = (const int*)d_in[8];    // [BS,N,N]
  float* out = (float*)d_out;

  // Workspace layout (all 16B-aligned):
  //   invdeg (64KB) | wpack (768KB) | featp (4MB) | agg0 (4MB) | agg1 (4MB)
  //   | h (4MB) | mpack (16MB)   total ~32.8MB
  char* ws = (char*)d_ws;
  const size_t FEAT_B = (size_t)BSZ * NN * DDIM * sizeof(__bf16);  // 4 MB
  const size_t WPK_B  = (size_t)6 * KT8 * NT * 32 * 16 * sizeof(__bf16);
  float*  invdeg = (float*)ws;
  __bf16* wpack  = (__bf16*)(ws + 65536);
  __bf16* featp  = (__bf16*)(ws + 65536 + WPK_B);
  __bf16* agg0   = (__bf16*)(ws + 65536 + WPK_B + FEAT_B);
  __bf16* agg1   = (__bf16*)(ws + 65536 + WPK_B + 2 * FEAT_B);
  __bf16* h      = (__bf16*)(ws + 65536 + WPK_B + 3 * FEAT_B);
  __bf16* mpack  = (__bf16*)(ws + 65536 + WPK_B + 4 * FEAT_B);

  const size_t WMAT = (size_t)KT8 * NT * 32 * 16;  // halves per packed matrix

  // One-time packs (masks/degrees are layer-invariant)
  rgcn_deg_kernel<<<(BSZ * NN + 255) / 256, 256, 0, stream>>>(aug, pun, invdeg);
  rgcn_maskpack_kernel<<<BSZ * JT * KT32, 32, 0, stream>>>(aug, pun, mpack);
  rgcn_wpack_kernel<<<6 * KT8 * NT, 32, 0, stream>>>(
      w_root1, w_rel1, w_rel1 + (size_t)DDIM * DDIM,
      w_root2, w_rel2, w_rel2 + (size_t)DDIM * DDIM, wpack);

  const int aggBlocks = BSZ * JT * (NT / 4);   // 2048 wave-tiles (16x64 each)
  const int cmbBlocks = BSZ * JT * (NT / 4);

  // Layer 1
  rgcn_featpack_kernel<true><<<BSZ * KT32 * NT, 32, 0, stream>>>(x, featp);
  rgcn_agg_kernel<<<aggBlocks, 32, 0, stream>>>(featp, mpack, invdeg, agg0, agg1);
  rgcn_combine_kernel<true, false><<<cmbBlocks, 32, 0, stream>>>(
      x, agg0, agg1, wpack, b1, h);

  // Layer 2 (reuses featp/agg buffers)
  rgcn_featpack_kernel<false><<<BSZ * KT32 * NT, 32, 0, stream>>>(h, featp);
  rgcn_agg_kernel<<<aggBlocks, 32, 0, stream>>>(featp, mpack, invdeg, agg0, agg1);
  rgcn_combine_kernel<false, true><<<cmbBlocks, 32, 0, stream>>>(
      h, agg0, agg1, wpack + 3 * WMAT, b2, out);
}
